// VQEmbeddingEMA_23811298689882
// MI455X (gfx1250) — matrix operand order repaired and checked by reading, not compiled, not run on hardware
//
#include <hip/hip_runtime.h>
#include <hip/hip_bf16.h>

// ---- problem constants (from reference) ----
#define DECAYF  0.999f
#define COMMITF 0.25f
#define EPSF    1e-5f

#define Bc   32
#define Hc   32
#define Wc   32
#define Nc   8
#define Mc   512
#define Dc   32
#define Cc   256       // N*D
#define HWc  1024      // H*W
#define CHWc 262144    // C*H*W
#define Lc   32768     // B*H*W

#define CHUNK   2048
#define PASS_L  128
#define NPASS   (CHUNK / PASS_L)   // 16
#define NCHUNK  (Lc / CHUNK)       // 16

// packed embedding row stride (u32): [16 hi pairs][16 lo pairs][e2][3 pad]
#define EPK_STRIDE 36
// x tile in LDS: [32 d rows][129 floats]  (TDM pads 1 DWORD per 128 DWORDs)
#define XROW 129

// LDS float offsets
#define EPK_F   0
#define DW_F    (Mc*EPK_STRIDE)                 // 18432
#define CNT_F   (DW_F + Mc*33)                  // 35328
#define XS_F    (CNT_F + Mc)                    // 35840
#define IDX_F   (XS_F + Dc*XROW)                // 39968
#define SMEM_F  (IDX_F + PASS_L)                // 40096 floats = 160384 B
#define XS_BYTE (XS_F * 4)                      // 143360

// ---- output layout (flat, reference return order) ----
#define OUT_OUT  0
#define OUT_LOSS 8388608
#define OUT_PPL  8388609
#define OUT_EMB  8388610
#define OUT_CNT  (OUT_EMB + Nc*Mc*Dc)   // 8519682
#define OUT_W    (OUT_CNT + Nc*Mc)      // 8523778

typedef __attribute__((ext_vector_type(16))) __bf16        v16bf;
typedef __attribute__((ext_vector_type(8)))  float         v8f;
typedef __attribute__((ext_vector_type(8)))  unsigned int  v8u;
typedef __attribute__((ext_vector_type(4)))  unsigned int  u32x4;
typedef __attribute__((ext_vector_type(8)))  int           i32x8;
typedef __attribute__((ext_vector_type(4)))  int           i32x4;

// ------------------------------------------------------------------
// Kernel 0: zero workspace accumulators (+ perplexity accumulator slot)
// ------------------------------------------------------------------
__global__ void vq_init(float* __restrict__ ws, float* __restrict__ out) {
  const int total = Nc*Mc + Nc*Mc*Dc + 1;   // counts + dw + loss scalar
  int i = blockIdx.x * 256 + threadIdx.x;
  if (i < total) ws[i] = 0.0f;
  if (i == 0) out[OUT_PPL] = 0.0f;
}

// ------------------------------------------------------------------
// Kernel 1: TDM-staged x tiles, bf16-split WMMA distances, branchless
//           argmin, quantized output, LDS counts/dw, loss partial.
// grid = (NCHUNK, Nc), block = 256 (8 wave32)
// ------------------------------------------------------------------
__global__ __launch_bounds__(256) void vq_main(
    const float* __restrict__ x, const float* __restrict__ emb,
    float* __restrict__ out, float* __restrict__ g_cnt,
    float* __restrict__ g_dw, float* __restrict__ g_loss)
{
  extern __shared__ float smem[];
  unsigned int* e_pk = (unsigned int*)smem;   // 512*36 u32  = 73728 B
  float* dw_s  = smem + DW_F;                 // 512*33 f    = 67584 B
  float* cnt_s = smem + CNT_F;                // 512 f
  float* x_s   = smem + XS_F;                 // 32*129 f (TDM-padded tile)
  int*   idx_s = (int*)(smem + IDX_F);        // 128

  const int tid  = threadIdx.x;
  const int n    = blockIdx.y;
  const int cb   = blockIdx.x * CHUNK;
  const int wv   = tid >> 5;
  const int lane = tid & 31;
  const int row  = lane & 15;
  const int half = lane >> 4;

  // ---- pack embedding[n] into bf16 hi/lo WMMA-B layout; zero accum ----
  for (int i = tid; i < Mc*Dc/2; i += 256) {       // 8192 pairs, b64 reads
    const int m = i >> 4, pr = i & 15;
    const float f0 = emb[n*Mc*Dc + m*Dc + 2*pr];
    const float f1 = emb[n*Mc*Dc + m*Dc + 2*pr + 1];
    const unsigned short h0 = __builtin_bit_cast(unsigned short, (__bf16)f0);
    const unsigned short h1 = __builtin_bit_cast(unsigned short, (__bf16)f1);
    const float r0 = f0 - __uint_as_float((unsigned)h0 << 16);
    const float r1 = f1 - __uint_as_float((unsigned)h1 << 16);
    const unsigned short l0 = __builtin_bit_cast(unsigned short, (__bf16)r0);
    const unsigned short l1 = __builtin_bit_cast(unsigned short, (__bf16)r1);
    e_pk[m*EPK_STRIDE + pr]      = ((unsigned)h1 << 16) | h0;
    e_pk[m*EPK_STRIDE + 16 + pr] = ((unsigned)l1 << 16) | l0;
  }
  for (int i = tid; i < Mc*Dc; i += 256) {
    int m = i >> 5, d = i & 31;
    dw_s[m*33 + d] = 0.0f;
  }
  for (int i = tid; i < Mc; i += 256) cnt_s[i] = 0.0f;
  __syncthreads();
  // e2 from the reconstructed values (consistent with dot precision)
  for (int m = tid; m < Mc; m += 256) {
    float s = 0.0f;
    #pragma unroll
    for (int pr = 0; pr < 16; ++pr) {
      const unsigned uh = e_pk[m*EPK_STRIDE + pr];
      const unsigned ul = e_pk[m*EPK_STRIDE + 16 + pr];
      const float v0 = __uint_as_float(uh << 16) + __uint_as_float(ul << 16);
      const float v1 = __uint_as_float(uh & 0xFFFF0000u) +
                       __uint_as_float(ul & 0xFFFF0000u);
      s += v0*v0 + v1*v1;
    }
    e_pk[m*EPK_STRIDE + 32] = __float_as_uint(s);
  }
  __syncthreads();

  float loss_acc = 0.0f;

  for (int p = 0; p < NPASS; ++p) {
    const int l0  = cb + p * PASS_L;
    const int b   = l0 >> 10;          // 128-l pass never crosses a b boundary
    const int hw0 = l0 & 1023;
    const float* xg = x + (size_t)b*CHWc + (size_t)(n*Dc)*HWc + hw0;

    // ---- TDM: DMA the 32x128 f32 tile (global row stride 1024) into
    //      x_s with 1-DWORD padding per 128 DWORDs (row stride 129). ----
    if (wv == 0) {
      const unsigned long long ga = (unsigned long long)(uintptr_t)xg;
      u32x4 g0;
      g0[0] = 1u;                                      // count=1 (valid D#)
      g0[1] = (unsigned)XS_BYTE;                       // lds_addr
      g0[2] = (unsigned)ga;                            // global_addr[31:0]
      g0[3] = (unsigned)((ga >> 32) & 0x01FFFFFFu) | (2u << 30); // addr hi|type=2
      i32x8 g1;
      g1[0] = (int)((2u << 16) | (1u << 20) | (6u << 22)); // 4B elem, pad 1DW/128DW
      g1[1] = (int)((PASS_L & 0xFFFFu) << 16);         // tensor_dim0 lo
      g1[2] = (int)((unsigned)Dc << 16);               // dim0 hi=0 | tensor_dim1
      g1[3] = (int)((unsigned)PASS_L << 16);           // dim1 hi=0 | tile_dim0
      g1[4] = (int)(unsigned)Dc;                       // tile_dim1 | tile_dim2=0
      g1[5] = (int)(unsigned)HWc;                      // tensor_dim0_stride lo
      g1[6] = 0; g1[7] = 0;
      i32x4 gz4 = {0, 0, 0, 0};
      i32x8 gz8 = {0, 0, 0, 0, 0, 0, 0, 0};
      __builtin_amdgcn_tensor_load_to_lds(g0, g1, gz4, gz4, gz8, 0);
      __builtin_amdgcn_s_wait_tensorcnt(0);
    }
    __syncthreads();

    // ---- A fragment (16 l-rows of this wave), bf16 hi/lo split ----
    v16bf a_hi, a_lo;
    {
      const float* xc = x_s + (wv*16 + row);   // column l, stride XROW per d
      #pragma unroll
      for (int v = 0; v < 8; ++v) {
        // contiguous k order used here (same as B): K = 16*half + 2*v + {0,1}
        const int k0 = (half << 4) | (v << 1);
        const float f0 = xc[k0 * XROW], f1 = xc[(k0 + 1) * XROW];
        const __bf16 h0 = (__bf16)f0, h1 = (__bf16)f1;
        a_hi[2*v]   = h0;  a_hi[2*v+1] = h1;
        a_lo[2*v]   = (__bf16)(f0 - (float)h0);
        a_lo[2*v+1] = (__bf16)(f1 - (float)h1);
      }
    }

    float best[8]; int bm[8];
    #pragma unroll
    for (int r = 0; r < 8; ++r) { best[r] = 3.4e38f; bm[r] = 0; }

    // double-buffered B fragments: 2x ds_load_b128 per hi / per lo
    const unsigned int* ep0 = e_pk + row*EPK_STRIDE + (half << 3);
    v8u bh = *(const v8u*)(ep0);
    v8u bl = *(const v8u*)(ep0 + 16);
    float e2m = __uint_as_float(e_pk[row*EPK_STRIDE + 32]);

    for (int mt = 0; mt < Mc/16; ++mt) {
      const int m  = mt*16 + row;            // this lane's m column
      const int mn = (mt < Mc/16 - 1) ? m + 16 : m;
      const unsigned int* epn = e_pk + mn*EPK_STRIDE + (half << 3);
      const v8u nbh = *(const v8u*)(epn);
      const v8u nbl = *(const v8u*)(epn + 16);
      const float ne2 = __uint_as_float(e_pk[mn*EPK_STRIDE + 32]);

      v8f acc = {0.f,0.f,0.f,0.f,0.f,0.f,0.f,0.f};
      acc = __builtin_amdgcn_wmma_f32_16x16x32_bf16(
                false, a_lo, false, __builtin_bit_cast(v16bf, bh),
                (short)0, acc, false, false);
      acc = __builtin_amdgcn_wmma_f32_16x16x32_bf16(
                false, a_hi, false, __builtin_bit_cast(v16bf, bl),
                (short)0, acc, false, false);
      acc = __builtin_amdgcn_wmma_f32_16x16x32_bf16(
                false, a_hi, false, __builtin_bit_cast(v16bf, bh),
                (short)0, acc, false, false);
      #pragma unroll
      for (int r = 0; r < 8; ++r) {
        // ||x||^2 constant across m -> argmin needs only e2 - 2*dot
        const float s = __builtin_fmaf(-2.0f, acc[r], e2m);
        const bool c = s < best[r];
        best[r] = c ? s : best[r];
        bm[r]   = c ? m : bm[r];
      }
      bh = nbh; bl = nbl; e2m = ne2;
    }

    // branchless argmin reduce across the 16 lanes of each half:
    // pack (monotonic score bits, m) into u64 and min-reduce.
    unsigned long long key[8];
    #pragma unroll
    for (int r = 0; r < 8; ++r) {
      unsigned u = __float_as_uint(best[r]);
      u ^= (unsigned)(((int)u) >> 31) | 0x80000000u;
      key[r] = ((unsigned long long)u << 32) | (unsigned)bm[r];
    }
    #pragma unroll
    for (int mask = 1; mask <= 8; mask <<= 1) {
      #pragma unroll
      for (int r = 0; r < 8; ++r) {
        const unsigned long long ok = __shfl_xor(key[r], mask, 32);
        key[r] = (ok < key[r]) ? ok : key[r];
      }
    }
    if (row == 0) {
      #pragma unroll
      for (int r = 0; r < 8; ++r)   // C/D row = r + 8*half
        idx_s[wv*16 + half*8 + r] = (int)(key[r] & 0xFFFFu);
    }
    __syncthreads();

    // ---- scatter / quantized output / loss (2 d-phases x 128 l) ----
    {
      const int lo  = tid & 127;
      const int dh  = tid >> 7;
      const int idx = idx_s[lo];
      if (dh == 0) atomicAdd(&cnt_s[idx], 1.0f);
      float* og = out + OUT_OUT + (size_t)b*CHWc + (size_t)(n*Dc)*HWc + hw0 + lo;
      #pragma unroll
      for (int d = dh; d < Dc; d += 2) {
        const int pr = d >> 1;
        const unsigned uh = e_pk[idx*EPK_STRIDE + pr];
        const unsigned ul = e_pk[idx*EPK_STRIDE + 16 + pr];
        const float fh = dh ? __uint_as_float(uh & 0xFFFF0000u)
                            : __uint_as_float(uh << 16);
        const float fl = dh ? __uint_as_float(ul & 0xFFFF0000u)
                            : __uint_as_float(ul << 16);
        const float q  = fh + fl;
        const float xv = x_s[d * XROW + lo];
        const float df = xv - q;
        loss_acc += df * df;
        atomicAdd(&dw_s[idx*33 + d], xv);
        og[(size_t)d * HWc] = q;             // straight-through value == q
      }
    }
    __syncthreads();
  }

  // flush LDS accumulators to global workspace
  for (int i = tid; i < Mc; i += 256) atomicAdd(&g_cnt[n*Mc + i], cnt_s[i]);
  for (int i = tid; i < Mc*Dc; i += 256) {
    int m = i >> 5, d = i & 31;
    atomicAdd(&g_dw[n*Mc*Dc + i], dw_s[m*33 + d]);
  }
  __syncthreads();
  // block loss reduction (reuse cnt_s region)
  float* red_s = cnt_s;
  red_s[tid] = loss_acc;
  __syncthreads();
  for (int s = 128; s > 0; s >>= 1) {
    if (tid < s) red_s[tid] += red_s[tid + s];
    __syncthreads();
  }
  if (tid == 0) atomicAdd(g_loss, red_s[0]);
}

// ------------------------------------------------------------------
// Kernel 2: EMA update, embedding_new, perplexity, loss. grid = Nc
// ------------------------------------------------------------------
__global__ __launch_bounds__(256) void vq_finalize(
    const float* __restrict__ g_cnt, const float* __restrict__ g_dw,
    const float* __restrict__ g_loss,
    const float* __restrict__ ema_count, const float* __restrict__ ema_weight,
    float* __restrict__ out)
{
  __shared__ float c_s[Mc];
  __shared__ float redA[256];
  __shared__ float redB[256];
  const int n = blockIdx.x;
  const int tid = threadIdx.x;

  float ptot = 0.0f, pent = 0.0f;
  for (int m = tid; m < Mc; m += 256) {
    float cnt = g_cnt[n*Mc + m];
    float cp  = DECAYF * ema_count[n*Mc + m] + (1.0f - DECAYF) * cnt;
    c_s[m] = cp;
    ptot  += cp;
    float avg = cnt * (1.0f / (float)Lc);
    pent  += avg * logf(avg + 1e-10f);
  }
  redA[tid] = ptot;
  redB[tid] = pent;
  __syncthreads();
  for (int s = 128; s > 0; s >>= 1) {
    if (tid < s) { redA[tid] += redA[tid + s]; redB[tid] += redB[tid + s]; }
    __syncthreads();
  }
  const float n_tot = redA[0];

  for (int m = tid; m < Mc; m += 256) {
    float cnew = (c_s[m] + EPSF) / (n_tot + (float)Mc * EPSF) * n_tot;
    out[OUT_CNT + n*Mc + m] = cnew;
    c_s[m] = cnew;
  }
  __syncthreads();

  for (int i = tid; i < Mc*Dc; i += 256) {
    int m = i >> 5;
    float wn = DECAYF * ema_weight[n*Mc*Dc + i] + (1.0f - DECAYF) * g_dw[n*Mc*Dc + i];
    out[OUT_W   + n*Mc*Dc + i] = wn;
    out[OUT_EMB + n*Mc*Dc + i] = wn / c_s[m];
  }

  if (tid == 0) {
    atomicAdd(&out[OUT_PPL], expf(-redB[0]));
    if (n == 0)
      out[OUT_LOSS] = COMMITF * g_loss[0] *
                      (1.0f / ((float)Nc * (float)Lc * (float)Dc));
  }
}

// ------------------------------------------------------------------
extern "C" void kernel_launch(void* const* d_in, const int* in_sizes, int n_in,
                              void* d_out, int out_size, void* d_ws, size_t ws_size,
                              hipStream_t stream) {
  (void)in_sizes; (void)n_in; (void)out_size; (void)ws_size;
  const float* x          = (const float*)d_in[0];
  const float* emb        = (const float*)d_in[1];
  const float* ema_count  = (const float*)d_in[2];
  const float* ema_weight = (const float*)d_in[3];
  float* out = (float*)d_out;
  float* ws  = (float*)d_ws;

  float* g_cnt  = ws;                       // N*M
  float* g_dw   = ws + Nc*Mc;               // N*M*D
  float* g_loss = ws + Nc*Mc + Nc*Mc*Dc;    // 1

  const int zn = Nc*Mc + Nc*Mc*Dc + 1;
  vq_init<<<(zn + 255) / 256, 256, 0, stream>>>(ws, out);

  const size_t smem_bytes = (size_t)SMEM_F * sizeof(float);
  dim3 grid(NCHUNK, Nc);
  vq_main<<<grid, 256, smem_bytes, stream>>>(x, emb, out, g_cnt, g_dw, g_loss);

  vq_finalize<<<Nc, 256, 0, stream>>>(g_cnt, g_dw, g_loss,
                                      ema_count, ema_weight, out);
}
